// attention_net_67894843015495
// MI455X (gfx1250) — matrix operand order repaired and checked by reading, not compiled, not run on hardware
//
#include <hip/hip_runtime.h>

// ---------------------------------------------------------------------------
// Types / helpers
// ---------------------------------------------------------------------------
typedef __attribute__((ext_vector_type(16))) __bf16 v16bf;
typedef __attribute__((ext_vector_type(8)))  __bf16 v8bf;
typedef __attribute__((ext_vector_type(8)))  float  v8f;

__device__ __forceinline__ unsigned short f2bf_bits(float f) {
  union { float f; unsigned u; } x; x.f = f;
  unsigned r = x.u + 0x7FFFu + ((x.u >> 16) & 1u);   // round-to-nearest-even
  return (unsigned short)(r >> 16);
}
__device__ __forceinline__ __bf16 bf_from_bits(unsigned short u) {
  union { unsigned short u; __bf16 h; } c; c.u = u; return c.h;
}

// ---------------------------------------------------------------------------
// f32 -> bf16 bulk conversion (one bandwidth pass per layer input)
// ---------------------------------------------------------------------------
__global__ void f32_to_bf16_kernel(const float* __restrict__ in,
                                   unsigned short* __restrict__ out, int total) {
  int i = blockIdx.x * blockDim.x + threadIdx.x;
  if (i < total) out[i] = f2bf_bits(in[i]);
}

// ---------------------------------------------------------------------------
// Global average pool over trailing HW:  out[bc] = mean(x[bc, :])
// ---------------------------------------------------------------------------
__global__ void gap_kernel(const float* __restrict__ x, float* __restrict__ out,
                           int BC, int HW) {
  int idx = blockIdx.x * blockDim.x + threadIdx.x;
  if (idx >= BC) return;
  const float* p = x + (long)idx * HW;
  float s = 0.f;
  for (int i = 0; i < HW; ++i) s += p[i];
  out[idx] = s / (float)HW;
}

// ---------------------------------------------------------------------------
// attention2d: relu(pooled @ fc1^T + b1) @ fc2^T + b2 -> softmax over K=4
// one wave per sample
// ---------------------------------------------------------------------------
__global__ __launch_bounds__(32)
void attention_kernel(const float* __restrict__ pooled,  // [B, Cin]
                      const float* __restrict__ fc1_w,   // [4, Cin]
                      const float* __restrict__ fc1_b,   // [4]
                      const float* __restrict__ fc2_w,   // [4, 4]
                      const float* __restrict__ fc2_b,   // [4]
                      float* __restrict__ att, int Cin) {
  int b = blockIdx.x;
  int lane = threadIdx.x;
  const float* pr = pooled + (long)b * Cin;
  float s0 = 0.f, s1 = 0.f, s2 = 0.f, s3 = 0.f;
  for (int c = lane; c < Cin; c += 32) {
    float pv = pr[c];
    s0 += pv * fc1_w[0 * Cin + c];
    s1 += pv * fc1_w[1 * Cin + c];
    s2 += pv * fc1_w[2 * Cin + c];
    s3 += pv * fc1_w[3 * Cin + c];
  }
  for (int off = 16; off > 0; off >>= 1) {
    s0 += __shfl_xor(s0, off, 32);
    s1 += __shfl_xor(s1, off, 32);
    s2 += __shfl_xor(s2, off, 32);
    s3 += __shfl_xor(s3, off, 32);
  }
  if (lane == 0) {
    float a0 = fmaxf(s0 + fc1_b[0], 0.f);
    float a1 = fmaxf(s1 + fc1_b[1], 0.f);
    float a2 = fmaxf(s2 + fc1_b[2], 0.f);
    float a3 = fmaxf(s3 + fc1_b[3], 0.f);
    float z[4];
    for (int k = 0; k < 4; ++k)
      z[k] = a0 * fc2_w[k * 4 + 0] + a1 * fc2_w[k * 4 + 1] +
             a2 * fc2_w[k * 4 + 2] + a3 * fc2_w[k * 4 + 3] + fc2_b[k];
    float mx = fmaxf(fmaxf(z[0], z[1]), fmaxf(z[2], z[3]));
    float e0 = __expf(z[0] - mx), e1 = __expf(z[1] - mx);
    float e2 = __expf(z[2] - mx), e3 = __expf(z[3] - mx);
    float inv = 1.f / (e0 + e1 + e2 + e3);
    att[b * 4 + 0] = e0 * inv; att[b * 4 + 1] = e1 * inv;
    att[b * 4 + 2] = e2 * inv; att[b * 4 + 3] = e3 * inv;
  }
}

// ---------------------------------------------------------------------------
// Per-sample expert-kernel mixing, stored as bf16:
//   aggw[b, j] = sum_k att[b,k] * w[k, j]
// (biggest layer: 144 MB bf16 -> resident in MI455X's 192 MB L2 for the conv)
// ---------------------------------------------------------------------------
__global__ void mix_weights_kernel(const float* __restrict__ w,       // [4, CC]
                                   const float* __restrict__ att,     // [B, 4]
                                   unsigned short* __restrict__ aggw, // [B, CC]
                                   int CC, int total) {
  int idx = blockIdx.x * blockDim.x + threadIdx.x;
  if (idx >= total) return;
  int b = idx / CC;
  int j = idx - b * CC;
  float v = att[b * 4 + 0] * w[j] +
            att[b * 4 + 1] * w[(long)CC + j] +
            att[b * 4 + 2] * w[2L * CC + j] +
            att[b * 4 + 3] * w[3L * CC + j];
  aggw[idx] = f2bf_bits(v);
}

// ---------------------------------------------------------------------------
// Implicit-GEMM per-sample conv using V_WMMA_F32_16X16X32_BF16.
// Grid: (ceil(N/32), ceil(Cout/64), B), block = 128 threads (4 waves).
// Block tile: 64(M) x 32(N). Each wave owns one 16-row M tile and runs TWO
// WMMAs per K-step (N columns 0..15 and 16..31) against one shared A fragment.
// The [32K x 32N] im2col B tile is staged cooperatively in double-buffered LDS.
//
// Per-thread staging trick: with blockDim=128 and K-step 32,
//   bk = (tid + 128*f) & 31 = tid & 31  (constant), bn = (tid>>5) + 4*f,
// so all N-side math (pn/Wo division, ih0/iw0) is hoisted out of the K-loop,
// and each K-step needs exactly ONE k->(c,r,s) decomposition per thread
// (cheap: KS is a template constant).
//
// Fragment fetches are fully vectorized:
//   A: 2x global b128 per lane  (K = 8*half + e / 16 + 8*half + e contiguous)
//   B: 4x ds b128 per lane      (K = 8*half + e / 16 + 8*half + e in the LDS row)
// C/D layout: lane n = L%16, vgpr r holds M = r + 8*(L/16).
// ---------------------------------------------------------------------------
template <int KS>
__global__ __launch_bounds__(128)
void dyn_conv_wmma_kernel(const unsigned short* __restrict__ xbf, // [B,Cin,H,W] bf16
                          const unsigned short* __restrict__ aggw,// [B,Cout,Ckk] bf16
                          const float* __restrict__ att,          // [B,4]
                          const float* __restrict__ bexp,         // [4,Cout]
                          float* __restrict__ y,                  // [B,Cout,Ho,Wo]
                          int Cin, int H, int W, int Cout,
                          int stride, int pad, int Ho, int Wo) {
  constexpr int KS2 = KS * KS;
  const int b    = blockIdx.z;
  const int n0   = blockIdx.x << 5;
  const int wid  = threadIdx.x >> 5;
  const int m0   = (blockIdx.y << 6) + (wid << 4);
  const int lane = threadIdx.x & 31;
  const int half = lane >> 4;
  const int l16  = lane & 15;
  const int tid  = threadIdx.x;

  const int N   = Ho * Wo;
  const int HW  = H * W;
  const int Ckk = Cin * KS2;

  // [buf][n][k] : row stride 40 bf16 = 80 B (16B aligned, bank-spread)
  __shared__ __bf16 bt[2][32][40];

  // ---- A row pointer (row clamped so OOB m still reads in-bounds) ----
  const int m  = m0 + l16;
  const int mc = m < Cout ? m : Cout - 1;
  const unsigned short* wrow = aggw + ((long)b * Cout + mc) * Ckk;

  const unsigned short* xb = xbf + (long)b * Cin * HW;

  // ---- hoisted N-side staging coordinates (bk/bn are kk-invariant) ----
  const int bk      = tid & 31;       // K offset within step (constant)
  const int bn_base = tid >> 5;       // bn = bn_base + 4*f, f = 0..7
  int  pih0[8], piw0[8];
  bool pnv[8];
  #pragma unroll
  for (int f = 0; f < 8; ++f) {
    int nn  = n0 + bn_base + (f << 2);
    bool v  = nn < N;
    int pn  = v ? nn : N - 1;
    int poh = pn / Wo, pow_ = pn - poh * Wo;
    pih0[f] = poh * stride - pad;
    piw0[f] = pow_ * stride - pad;
    pnv[f]  = v;
  }

  // ---- per-lane output columns (two N tiles) ----
  const int nA = n0 + l16, nB = nA + 16;
  const bool nvA = nA < N, nvB = nB < N;
  const int ncA = nvA ? nA : N - 1;
  const int ncB = nvB ? nB : N - 1;
  const int ohA = ncA / Wo, owA = ncA - ohA * Wo;
  const int ohB = ncB / Wo, owB = ncB - ohB * Wo;

  v8f acc0 = {}, acc1 = {};
  int buf = 0;
  for (int kk = 0; kk < Ckk; kk += 32) {
    // ---- cooperative staging: one (c,r,s) decomposition per thread ----
    {
      int k   = kk + bk;
      int c   = k / KS2;
      int rem = k - c * KS2;
      int r   = rem / KS;
      int s   = rem - r * KS;
      const unsigned short* xrow = xb + (long)c * HW;
      #pragma unroll
      for (int f = 0; f < 8; ++f) {
        int ih = pih0[f] + r, iw = piw0[f] + s;
        unsigned short v = 0;
        if (pnv[f] && (unsigned)ih < (unsigned)H && (unsigned)iw < (unsigned)W)
          v = xrow[ih * W + iw];
        bt[buf][bn_base + (f << 2)][bk] = bf_from_bits(v);
      }
    }
    __syncthreads();

    // ---- vectorized fragment fetch + 2 WMMAs sharing the A fragment ----
    v8bf alo = *(const v8bf*)(wrow + kk + 8 * half);
    v8bf ahi = *(const v8bf*)(wrow + kk + 16 + 8 * half);
    v16bf av = __builtin_shufflevector(alo, ahi, 0, 1, 2, 3, 4, 5, 6, 7,
                                       8, 9, 10, 11, 12, 13, 14, 15);

    const __bf16* brow0 = &bt[buf][l16][8 * half];
    const __bf16* brow1 = &bt[buf][16 + l16][8 * half];
    v8bf b0lo = *(const v8bf*)brow0;
    v8bf b0hi = *(const v8bf*)(brow0 + 16);
    v8bf b1lo = *(const v8bf*)brow1;
    v8bf b1hi = *(const v8bf*)(brow1 + 16);
    v16bf bv0 = __builtin_shufflevector(b0lo, b0hi, 0, 1, 2, 3, 4, 5, 6, 7,
                                        8, 9, 10, 11, 12, 13, 14, 15);
    v16bf bv1 = __builtin_shufflevector(b1lo, b1hi, 0, 1, 2, 3, 4, 5, 6, 7,
                                        8, 9, 10, 11, 12, 13, 14, 15);

    acc0 = __builtin_amdgcn_wmma_f32_16x16x32_bf16(
        false, av, false, bv0, (short)0, acc0, false, false);
    acc1 = __builtin_amdgcn_wmma_f32_16x16x32_bf16(
        false, av, false, bv1, (short)0, acc1, false, false);
    buf ^= 1;   // double buffer: safe with one barrier per iteration
  }

  // ---- epilogue: attention-mixed bias + stores for both N tiles ----
  const float a0 = att[b * 4 + 0], a1 = att[b * 4 + 1];
  const float a2 = att[b * 4 + 2], a3 = att[b * 4 + 3];
  #pragma unroll
  for (int r = 0; r < 8; ++r) {
    int mm = m0 + r + 8 * half;
    if (mm < Cout) {
      float bias = a0 * bexp[0 * Cout + mm] + a1 * bexp[1 * Cout + mm] +
                   a2 * bexp[2 * Cout + mm] + a3 * bexp[3 * Cout + mm];
      long row = ((long)b * Cout + mm) * Ho;
      if (nvA) y[(row + ohA) * Wo + owA] = acc0[r] + bias;
      if (nvB) y[(row + ohB) * Wo + owB] = acc1[r] + bias;
    }
  }
}

// ---------------------------------------------------------------------------
// Training-mode BatchNorm2d (batch stats, biased var), in place.
// ---------------------------------------------------------------------------
__global__ __launch_bounds__(256)
void bn_kernel(float* __restrict__ y, const float* __restrict__ gamma,
               const float* __restrict__ beta, int B, int C, int HW) {
  int c = blockIdx.x;
  int t = threadIdx.x;
  __shared__ float ssum[256], ssq[256];
  int n = B * HW;
  float s = 0.f, q = 0.f;
  for (int i = t; i < n; i += 256) {
    int b = i / HW, p = i - b * HW;
    float v = y[((long)b * C + c) * HW + p];
    s += v; q += v * v;
  }
  ssum[t] = s; ssq[t] = q;
  __syncthreads();
  for (int off = 128; off > 0; off >>= 1) {
    if (t < off) { ssum[t] += ssum[t + off]; ssq[t] += ssq[t + off]; }
    __syncthreads();
  }
  float mean = ssum[0] / (float)n;
  float var  = ssq[0] / (float)n - mean * mean;
  float inv  = rsqrtf(var + 1e-5f);
  float g  = gamma[c] * inv;
  float bb = beta[c] - mean * g;
  for (int i = t; i < n; i += 256) {
    int b = i / HW, p = i - b * HW;
    long o = ((long)b * C + c) * HW + p;
    y[o] = y[o] * g + bb;
  }
}

// ---------------------------------------------------------------------------
// Final FC + sigmoid over concat([g1(64), g2(128), g3(32)]). One wave / sample.
// ---------------------------------------------------------------------------
__global__ __launch_bounds__(32)
void final_kernel(const float* __restrict__ g1, const float* __restrict__ g2,
                  const float* __restrict__ g3, const float* __restrict__ fc_w,
                  const float* __restrict__ fc_b, float* __restrict__ out) {
  int b = blockIdx.x;
  int lane = threadIdx.x;
  float s = 0.f;
  for (int j = lane; j < 64;  j += 32) s += g1[b * 64 + j]  * fc_w[j];
  for (int j = lane; j < 128; j += 32) s += g2[b * 128 + j] * fc_w[64 + j];
  for (int j = lane; j < 32;  j += 32) s += g3[b * 32 + j]  * fc_w[192 + j];
  for (int off = 16; off > 0; off >>= 1) s += __shfl_xor(s, off, 32);
  if (lane == 0) out[b] = 1.f / (1.f + __expf(-(s + fc_b[0])));
}

// ---------------------------------------------------------------------------
// Host side
// ---------------------------------------------------------------------------
namespace {

struct LayerCfg { int Cin, Cout, ks, stride, pad; };

void run_dyn_conv(hipStream_t stream, const float* xin, unsigned short* xbf,
                  float* yout, int B, int Cin, int H, int W,
                  const float* fc1w, const float* fc1b,
                  const float* fc2w, const float* fc2b,
                  const float* w, const float* bexp,
                  int Cout, int ks, int stride, int pad,
                  float* pooled, float* att, unsigned short* aggw,
                  int& HoOut, int& WoOut) {
  int Ho = (H + 2 * pad - ks) / stride + 1;
  int Wo = (W + 2 * pad - ks) / stride + 1;
  int HW = H * W;
  int BC = B * Cin;
  gap_kernel<<<(BC + 255) / 256, 256, 0, stream>>>(xin, pooled, BC, HW);
  attention_kernel<<<B, 32, 0, stream>>>(pooled, fc1w, fc1b, fc2w, fc2b, att, Cin);
  int CC = Cout * Cin * ks * ks;
  int total = B * CC;
  mix_weights_kernel<<<(total + 255) / 256, 256, 0, stream>>>(w, att, aggw, CC, total);
  int xtot = B * Cin * HW;
  f32_to_bf16_kernel<<<(xtot + 255) / 256, 256, 0, stream>>>(xin, xbf, xtot);
  int N = Ho * Wo;
  dim3 grid((N + 31) / 32, (Cout + 63) / 64, B);
  switch (ks) {
    case 2:
      dyn_conv_wmma_kernel<2><<<grid, 128, 0, stream>>>(
          xbf, aggw, att, bexp, yout, Cin, H, W, Cout, stride, pad, Ho, Wo);
      break;
    case 3:
      dyn_conv_wmma_kernel<3><<<grid, 128, 0, stream>>>(
          xbf, aggw, att, bexp, yout, Cin, H, W, Cout, stride, pad, Ho, Wo);
      break;
    default:
      dyn_conv_wmma_kernel<8><<<grid, 128, 0, stream>>>(
          xbf, aggw, att, bexp, yout, Cin, H, W, Cout, stride, pad, Ho, Wo);
      break;
  }
  HoOut = Ho; WoOut = Wo;
}

} // namespace

extern "C" void kernel_launch(void* const* d_in, const int* in_sizes, int n_in,
                              void* d_out, int out_size, void* d_ws, size_t ws_size,
                              hipStream_t stream) {
  (void)n_in; (void)out_size; (void)ws_size;
  const int B = 64;

  const float* P[80];
  for (int i = 0; i < n_in && i < 80; ++i) P[i] = (const float*)d_in[i];

  const float *x, *fcw, *fcb;
  const float *c_fc1w[10], *c_fc1b[10], *c_fc2w[10], *c_fc2b[10], *c_w[10], *c_b[10];
  const float *bn_g[6], *bn_b[6];

  // Resolve flattening order deterministically from input shapes:
  // insertion-order DFS puts x (6,422,528 elems) first; jax-pytree sorted
  // order puts bn11.beta (256 elems) first and x last.
  bool insertion = (in_sizes[0] == 64 * 512 * 14 * 14);
  if (insertion) {
    x = P[0];
    for (int i = 0; i < 10; ++i) {
      int base = 1 + 6 * i;
      c_fc1w[i] = P[base + 0]; c_fc1b[i] = P[base + 1];
      c_fc2w[i] = P[base + 2]; c_fc2b[i] = P[base + 3];
      c_w[i]    = P[base + 4]; c_b[i]    = P[base + 5];
    }
    for (int j = 0; j < 6; ++j) { bn_g[j] = P[61 + 2 * j]; bn_b[j] = P[62 + 2 * j]; }
    fcw = P[73]; fcb = P[74];
  } else {
    // sorted: bn*(beta,gamma) x6, conv*(b,fc1_b,fc1_w,fc2_b,fc2_w,w) x10, fc_b, fc_w, x
    for (int j = 0; j < 6; ++j) { bn_b[j] = P[2 * j]; bn_g[j] = P[2 * j + 1]; }
    for (int i = 0; i < 10; ++i) {
      int base = 12 + 6 * i;
      c_b[i]    = P[base + 0]; c_fc1b[i] = P[base + 1];
      c_fc1w[i] = P[base + 2]; c_fc2b[i] = P[base + 3];
      c_fc2w[i] = P[base + 4]; c_w[i]    = P[base + 5];
    }
    fcb = P[72]; fcw = P[73]; x = P[74];
  }

  // layer configs: c11..c14, c21..c23, c31..c33
  const LayerCfg L[10] = {
      {512, 256, 3, 1, 1}, {256, 128, 3, 1, 1}, {128, 128, 8, 1, 1}, {128, 64, 2, 1, 0},
      {512, 256, 3, 2, 1}, {256, 128, 3, 2, 1}, {128, 128, 2, 1, 0},
      {512, 128, 3, 1, 0}, {128, 64, 3, 1, 0},  {64, 32, 3, 1, 0}};

  // workspace bump allocator
  char* ws = (char*)d_ws;
  auto alloc = [&](size_t bytes) -> void* {
    void* p = (void*)ws;
    ws += (bytes + 255) & ~(size_t)255;
    return p;
  };
  float* pooled = (float*)alloc((size_t)B * 512 * sizeof(float));
  float* att    = (float*)alloc((size_t)B * 4 * sizeof(float));
  unsigned short* aggw =
      (unsigned short*)alloc((size_t)B * 256 * 4608 * sizeof(unsigned short)); // 144 MB (c11 max, fits L2)
  unsigned short* xbf =
      (unsigned short*)alloc((size_t)B * 512 * 196 * sizeof(unsigned short));  // bf16 input scratch
  float* actA = (float*)alloc((size_t)B * 256 * 196 * sizeof(float));
  float* actB = (float*)alloc((size_t)B * 256 * 196 * sizeof(float));
  float* g1 = (float*)alloc((size_t)B * 64 * sizeof(float));
  float* g2 = (float*)alloc((size_t)B * 128 * sizeof(float));
  float* g3 = (float*)alloc((size_t)B * 32 * sizeof(float));

  int Ho, Wo;

  // ---------------- branch 1 ----------------
  run_dyn_conv(stream, x, xbf, actA, B, L[0].Cin, 14, 14, c_fc1w[0], c_fc1b[0],
               c_fc2w[0], c_fc2b[0], c_w[0], c_b[0], L[0].Cout, L[0].ks, L[0].stride,
               L[0].pad, pooled, att, aggw, Ho, Wo);             // [256,14,14]
  bn_kernel<<<256, 256, 0, stream>>>(actA, bn_g[0], bn_b[0], B, 256, Ho * Wo);
  run_dyn_conv(stream, actA, xbf, actB, B, L[1].Cin, Ho, Wo, c_fc1w[1], c_fc1b[1],
               c_fc2w[1], c_fc2b[1], c_w[1], c_b[1], L[1].Cout, L[1].ks, L[1].stride,
               L[1].pad, pooled, att, aggw, Ho, Wo);             // [128,14,14]
  bn_kernel<<<128, 256, 0, stream>>>(actB, bn_g[1], bn_b[1], B, 128, Ho * Wo);
  run_dyn_conv(stream, actB, xbf, actA, B, L[2].Cin, Ho, Wo, c_fc1w[2], c_fc1b[2],
               c_fc2w[2], c_fc2b[2], c_w[2], c_b[2], L[2].Cout, L[2].ks, L[2].stride,
               L[2].pad, pooled, att, aggw, Ho, Wo);             // [128,9,9]
  run_dyn_conv(stream, actA, xbf, actB, B, L[3].Cin, Ho, Wo, c_fc1w[3], c_fc1b[3],
               c_fc2w[3], c_fc2b[3], c_w[3], c_b[3], L[3].Cout, L[3].ks, L[3].stride,
               L[3].pad, pooled, att, aggw, Ho, Wo);             // [64,8,8]
  gap_kernel<<<(B * 64 + 255) / 256, 256, 0, stream>>>(actB, g1, B * 64, Ho * Wo);

  // ---------------- branch 2 ----------------
  run_dyn_conv(stream, x, xbf, actA, B, L[4].Cin, 14, 14, c_fc1w[4], c_fc1b[4],
               c_fc2w[4], c_fc2b[4], c_w[4], c_b[4], L[4].Cout, L[4].ks, L[4].stride,
               L[4].pad, pooled, att, aggw, Ho, Wo);             // [256,7,7]
  bn_kernel<<<256, 256, 0, stream>>>(actA, bn_g[2], bn_b[2], B, 256, Ho * Wo);
  run_dyn_conv(stream, actA, xbf, actB, B, L[5].Cin, Ho, Wo, c_fc1w[5], c_fc1b[5],
               c_fc2w[5], c_fc2b[5], c_w[5], c_b[5], L[5].Cout, L[5].ks, L[5].stride,
               L[5].pad, pooled, att, aggw, Ho, Wo);             // [128,4,4]
  bn_kernel<<<128, 256, 0, stream>>>(actB, bn_g[3], bn_b[3], B, 128, Ho * Wo);
  run_dyn_conv(stream, actB, xbf, actA, B, L[6].Cin, Ho, Wo, c_fc1w[6], c_fc1b[6],
               c_fc2w[6], c_fc2b[6], c_w[6], c_b[6], L[6].Cout, L[6].ks, L[6].stride,
               L[6].pad, pooled, att, aggw, Ho, Wo);             // [128,3,3]
  gap_kernel<<<(B * 128 + 255) / 256, 256, 0, stream>>>(actA, g2, B * 128, Ho * Wo);

  // ---------------- branch 3 ----------------
  run_dyn_conv(stream, x, xbf, actA, B, L[7].Cin, 14, 14, c_fc1w[7], c_fc1b[7],
               c_fc2w[7], c_fc2b[7], c_w[7], c_b[7], L[7].Cout, L[7].ks, L[7].stride,
               L[7].pad, pooled, att, aggw, Ho, Wo);             // [128,12,12]
  bn_kernel<<<128, 256, 0, stream>>>(actA, bn_g[4], bn_b[4], B, 128, Ho * Wo);
  run_dyn_conv(stream, actA, xbf, actB, B, L[8].Cin, Ho, Wo, c_fc1w[8], c_fc1b[8],
               c_fc2w[8], c_fc2b[8], c_w[8], c_b[8], L[8].Cout, L[8].ks, L[8].stride,
               L[8].pad, pooled, att, aggw, Ho, Wo);             // [64,10,10]
  bn_kernel<<<64, 256, 0, stream>>>(actB, bn_g[5], bn_b[5], B, 64, Ho * Wo);
  run_dyn_conv(stream, actB, xbf, actA, B, L[9].Cin, Ho, Wo, c_fc1w[9], c_fc1b[9],
               c_fc2w[9], c_fc2b[9], c_w[9], c_b[9], L[9].Cout, L[9].ks, L[9].stride,
               L[9].pad, pooled, att, aggw, Ho, Wo);             // [32,8,8]
  gap_kernel<<<(B * 32 + 255) / 256, 256, 0, stream>>>(actA, g3, B * 32, Ho * Wo);

  // ---------------- head ----------------
  final_kernel<<<B, 32, 0, stream>>>(g1, g2, g3, fcw, fcb, (float*)d_out);
}
